// StateIntegratorND_66443144069469
// MI455X (gfx1250) — compile-verified, matrix operand order checked by reading, not run on hardware
//
#include <hip/hip_runtime.h>
#include <hip/hip_fp16.h>
#include <math.h>

typedef __attribute__((ext_vector_type(16))) _Float16 v16h;
typedef __attribute__((ext_vector_type(8)))  _Float16 v8h;
typedef __attribute__((ext_vector_type(8)))  float    v8f;
typedef unsigned int u32x4 __attribute__((ext_vector_type(4)));
typedef int          i32x4 __attribute__((ext_vector_type(4)));
typedef int          i32x8 __attribute__((ext_vector_type(8)));

#if defined(__has_builtin)
#  if __has_builtin(__builtin_amdgcn_tanhf)
#    define TANHF(x) __builtin_amdgcn_tanhf(x)
#  endif
#  if __has_builtin(__builtin_amdgcn_tensor_load_to_lds)
#    define HAVE_TDM 1
#  endif
#endif
#ifndef TANHF
#  define TANHF(x) tanhf(x)
#endif

#define WMMA16(a,b,c) __builtin_amdgcn_wmma_f32_16x16x32_f16(false,(a),false,(b),(short)0,(c),false,false)

#define DIN   20      // nx + nu
#define HC    128     // hidden
#define NXC   16

// fragment-packed weight sizes (halfs)
#define W1_HALFS (8*32*16)       // K=32(pad, incl bias row k=20) x N=128 -> 8 ntiles
#define W2_HALFS (4*8*32*16)     // K=128 x N=128 -> 4 kt x 8 nt
#define W3_HALFS (4*1*32*16)     // K=128 x N=16  -> 4 kt x 1 nt
#define WF_HALFS (W1_HALFS + W2_HALFS + W3_HALFS)   // 22528
#define BLOB_BYTES ((W1_HALFS + W2_HALFS) * 2)      // 40960 (staged to LDS)

// dopri5 coefficients pre-scaled by h = DT/N_STEPS
#define HSUB (0.01/4.0)
__device__ __constant__ const float kC21 = (float)(HSUB*(1.0/5.0));
__device__ __constant__ const float kC31 = (float)(HSUB*(3.0/40.0)),  kC32 = (float)(HSUB*(9.0/40.0));
__device__ __constant__ const float kC41 = (float)(HSUB*(44.0/45.0)), kC42 = (float)(HSUB*(-56.0/15.0)), kC43 = (float)(HSUB*(32.0/9.0));
__device__ __constant__ const float kC51 = (float)(HSUB*(19372.0/6561.0)), kC52 = (float)(HSUB*(-25360.0/2187.0)),
                                    kC53 = (float)(HSUB*(64448.0/6561.0)), kC54 = (float)(HSUB*(-212.0/729.0));
__device__ __constant__ const float kC61 = (float)(HSUB*(9017.0/3168.0)),  kC62 = (float)(HSUB*(-355.0/33.0)),
                                    kC63 = (float)(HSUB*(46732.0/5247.0)), kC64 = (float)(HSUB*(49.0/176.0)),
                                    kC65 = (float)(HSUB*(-5103.0/18656.0));
__device__ __constant__ const float kD1 = (float)(HSUB*(35.0/384.0)),  kD3 = (float)(HSUB*(500.0/1113.0)),
                                    kD4 = (float)(HSUB*(125.0/192.0)), kD5 = (float)(HSUB*(-2187.0/6784.0)),
                                    kD6 = (float)(HSUB*(11.0/84.0));

__device__ __forceinline__ v8f vfma8(v8f a, float s, v8f b) {
  v8f r;
#pragma unroll
  for (int i = 0; i < 8; ++i) r[i] = fmaf(s, b[i], a[i]);
  return r;
}

// ---------------------------------------------------------------------------
// Prep: pack f32 weights into f16 WMMA B-fragment layout in workspace.
// Fragment (kt,nt): 32 lanes x 16 halfs contiguous. Lane l (g=l>>4, n=nt*16+(l&15)):
//   half j<8  -> k = kt*32 + 8*g + j            (VGPRs 0..3)
//   half j>=8 -> k = kt*32 + 16 + 8*g + (j-8)   (VGPRs 4..7)
// Layer-1 bias is folded in as K-row 20 (paired with xu col 20 == 1.0).
// ---------------------------------------------------------------------------
__global__ void sin_prep_kernel(const float* __restrict__ W1, const float* __restrict__ W2,
                                const float* __restrict__ W3, const float* __restrict__ b1,
                                const float* __restrict__ b2, const float* __restrict__ b3,
                                _Float16* __restrict__ wf, float* __restrict__ wb) {
  const int t = threadIdx.x, n_thr = blockDim.x;
  for (int i = t; i < W1_HALFS; i += n_thr) {                 // kt = 0 only, K padded to 32
    int j = i & 15, l = (i >> 4) & 31, nt = i >> 9;
    int g = l >> 4, n = nt * 16 + (l & 15);
    int k = (j < 8) ? (8 * g + j) : (16 + 8 * g + (j - 8));
    float v = (k < DIN) ? W1[k * HC + n] : ((k == DIN) ? b1[n] : 0.f);
    wf[i] = (_Float16)v;
  }
  for (int i = t; i < W2_HALFS; i += n_thr) {
    int j = i & 15, l = (i >> 4) & 31, nt = (i >> 9) & 7, kt = i >> 12;
    int g = l >> 4, n = nt * 16 + (l & 15);
    int k = kt * 32 + ((j < 8) ? (8 * g + j) : (16 + 8 * g + (j - 8)));
    wf[W1_HALFS + i] = (_Float16)W2[k * HC + n];
  }
  for (int i = t; i < W3_HALFS; i += n_thr) {
    int j = i & 15, l = (i >> 4) & 31, kt = i >> 9;
    int g = l >> 4, n = (l & 15);
    int k = kt * 32 + ((j < 8) ? (8 * g + j) : (16 + 8 * g + (j - 8)));
    wf[W1_HALFS + W2_HALFS + i] = (_Float16)W3[k * NXC + n];
  }
  for (int i = t; i < HC; i += n_thr) { wb[i] = b1[i]; wb[HC + i] = b2[i]; }
  for (int i = t; i < NXC; i += n_thr) wb[2 * HC + i] = b3[i];
}

__global__ void sin_zero_kernel(float* o, int n) {
  int i = blockIdx.x * blockDim.x + threadIdx.x;
  if (i < n) o[i] = 0.f;
}

// ---------------------------------------------------------------------------
// Main integrator: 4 waves/block, each wave owns a 16-particle tile.
// Activation buffer per wave: 16 rows x 168 f16 (cols 0..127 = h, 136..167 = xu),
// row stride 336B -> 16B aligned, bank-conflict-free across rows.
// Weights staged into LDS once per block by the Tensor Data Mover (wave 0).
// ---------------------------------------------------------------------------
__global__ __launch_bounds__(128) void sin_integrate_kernel(
    const float* __restrict__ sp, const float* __restrict__ Wp,
    const _Float16* __restrict__ wf, const float* __restrict__ wb,
    float* __restrict__ out, int P, int S) {
  __shared__ __align__(32) unsigned char sBlob[BLOB_BYTES];   // W1f | W2f
  __shared__ __align__(16) _Float16 sHX[4][16 * 168];

  const int lane = threadIdx.x & 31;
  const int wave = threadIdx.x >> 5;
  const int g    = lane >> 4;        // half-wave group
  const int col  = lane & 15;        // N column (B/C/D layout) / row (A layout)
  const int tileBase = blockIdx.x * 64 + wave * 16;

  __builtin_prefetch(sp + (size_t)tileBase * DIN, 0, 1);      // global_prefetch_b8

  // --- stage fragment-packed W1/W2 into LDS ---
#if defined(HAVE_TDM)
  if (threadIdx.x < 32) {            // wave 0 drives the TDM; EXEC is ignored by it
    const unsigned int qw = BLOB_BYTES / 8;                   // 5120 x 8B elements
    const unsigned long long ga = (unsigned long long)(uintptr_t)wf;
    const unsigned int lds = (unsigned int)(uintptr_t)&sBlob[0];  // flat[31:0] == LDS addr
    u32x4 g0;
    g0[0] = 1u;                                               // count=1, no gather
    g0[1] = lds;                                              // lds_addr
    g0[2] = (unsigned int)ga;                                 // global_addr[31:0]
    g0[3] = (unsigned int)((ga >> 32) & 0x01FFFFFFu) | 0x80000000u;  // addr[56:32] | type=2
    i32x8 g1;
    g1[0] = 0x00030000;                                       // data_size = 8B
    g1[1] = (int)((qw & 0xFFFFu) << 16);                      // tensor_dim0 lo16
    g1[2] = (int)(((qw >> 16) & 0xFFFFu) | (1u << 16));       // dim0 hi16 | tensor_dim1=1
    g1[3] = (int)((qw & 0xFFFFu) << 16);                      // tile_dim0
    g1[4] = 1;                                                // tile_dim1=1
    g1[5] = (int)qw;                                          // tensor_dim0_stride lo32
    g1[6] = (int)((qw & 0xFFFFu) << 16);                      // tensor_dim1_stride lo16
    g1[7] = 0;
    i32x4 gz = {0, 0, 0, 0};
#if defined(__clang_major__) && __clang_major__ >= 23
    i32x8 gz8 = {0, 0, 0, 0, 0, 0, 0, 0};
    __builtin_amdgcn_tensor_load_to_lds(g0, g1, gz, gz, gz8, 0);
#else
    __builtin_amdgcn_tensor_load_to_lds(g0, g1, gz, gz, 0);
#endif
    __builtin_amdgcn_s_wait_tensorcnt(0);
  }
#else
  {
    const uint4* src = (const uint4*)wf;
    uint4* dst = (uint4*)sBlob;
    for (int i = threadIdx.x; i < BLOB_BYTES / 16; i += blockDim.x) dst[i] = src[i];
  }
#endif

  // biases -> registers (L2-resident global reads, once)
  float bias2[8];
#pragma unroll
  for (int nt = 0; nt < 8; ++nt) bias2[nt] = wb[HC + nt * 16 + col];
  const float bias3 = wb[2 * HC + col];

  __syncthreads();   // LDS weights visible to all waves

  _Float16* hx   = &sHX[wave][0];
  _Float16* rowb = hx + col * 168;   // this lane's A-layout row base

  const v16h* w1v = (const v16h*)sBlob;
  const v16h* w2v = (const v16h*)(sBlob + W1_HALFS * 2);
  const v16h* w3g = (const v16h*)(wf + W1_HALFS + W2_HALFS);  // stays in L2

  // init xu tail: u cols (16..19), bias-one col (20), zero pad (21..31); persists
  for (int idx = lane; idx < 256; idx += 32) {
    int r = idx >> 4, c = idx & 15;                       // xu col 16+c
    int p = tileBase + r; int pc = (p < P) ? p : (P - 1);
    float v = (c < 4) ? sp[pc * DIN + NXC + c] : ((c == 4) ? 1.0f : 0.f);
    hx[r * 168 + 152 + c] = (_Float16)v;
  }

  // initial state x in C/D layout: lane holds column `col` of rows (v + 8g)
  v8f x;
#pragma unroll
  for (int v = 0; v < 8; ++v) {
    int p = tileBase + v + 8 * g; int pc = (p < P) ? p : (P - 1);
    x[v] = sp[pc * DIN + col];
  }

  // A-fragment gather: two 16B runs per lane (K=8g..8g+7 and 16+8g..16+8g+7)
  auto ldA = [&](const _Float16* p) -> v16h {
    v8h lo = *(const v8h*)(p + 8 * g);
    v8h hi = *(const v8h*)(p + 16 + 8 * g);
    return __builtin_shufflevector(lo, hi, 0, 1, 2, 3, 4, 5, 6, 7,
                                           8, 9, 10, 11, 12, 13, 14, 15);
  };

  // one drift evaluation: y (C/D layout) -> k (C/D layout)
  auto feval = [&](v8f y) -> v8f {
    // scatter y into xu columns 136..151 (f16)
#pragma unroll
    for (int v = 0; v < 8; ++v) hx[(v + 8 * g) * 168 + 136 + col] = (_Float16)y[v];
    __syncthreads();

    // layer 1: [16x32] @ [32x128], bias folded into K-row 20, C = 0
    v8f acc[8];
    v16h a1 = ldA(rowb + 136);
#pragma unroll
    for (int nt = 0; nt < 8; ++nt) {
      v8f c = {};
      acc[nt] = WMMA16(a1, w1v[nt * 32 + lane], c);
    }
    __syncthreads();   // xu reads everywhere done before h overwrite below
#pragma unroll
    for (int nt = 0; nt < 8; ++nt)
#pragma unroll
      for (int v = 0; v < 8; ++v)
        hx[(v + 8 * g) * 168 + nt * 16 + col] = (_Float16)TANHF(acc[nt][v]);
    __syncthreads();

    // layer 2: [16x128] @ [128x128] — A fragments hoisted (loaded once per kt)
    v16h a2[4];
#pragma unroll
    for (int kt = 0; kt < 4; ++kt) a2[kt] = ldA(rowb + kt * 32);
#pragma unroll
    for (int nt = 0; nt < 8; ++nt) {
      v8f c;
#pragma unroll
      for (int i = 0; i < 8; ++i) c[i] = bias2[nt];
#pragma unroll
      for (int kt = 0; kt < 4; ++kt)
        c = WMMA16(a2[kt], w2v[(kt * 8 + nt) * 32 + lane], c);
      acc[nt] = c;
    }
    __syncthreads();   // all layer-2 reads complete before overwrite
#pragma unroll
    for (int nt = 0; nt < 8; ++nt)
#pragma unroll
      for (int v = 0; v < 8; ++v)
        hx[(v + 8 * g) * 168 + nt * 16 + col] = (_Float16)TANHF(acc[nt][v]);
    __syncthreads();

    // layer 3: [16x128] @ [128x16]
    v8f k;
#pragma unroll
    for (int i = 0; i < 8; ++i) k[i] = bias3;
#pragma unroll
    for (int kt = 0; kt < 4; ++kt)
      k = WMMA16(ldA(rowb + kt * 32), w3g[kt * 32 + lane], k);
    return k;
  };

  // fixed-step dopri5
  for (int step = 0; step < 4; ++step) {
    v8f k1 = feval(x);
    v8f y  = vfma8(x, kC21, k1);
    v8f k2 = feval(y);
    y = vfma8(vfma8(x, kC31, k1), kC32, k2);
    v8f k3 = feval(y);
    y = vfma8(vfma8(vfma8(x, kC41, k1), kC42, k2), kC43, k3);
    v8f k4 = feval(y);
    y = vfma8(vfma8(vfma8(vfma8(x, kC51, k1), kC52, k2), kC53, k3), kC54, k4);
    v8f k5 = feval(y);
    y = vfma8(vfma8(vfma8(vfma8(vfma8(x, kC61, k1), kC62, k2), kC63, k3), kC64, k4), kC65, k5);
    v8f k6 = feval(y);
    x = vfma8(vfma8(vfma8(vfma8(vfma8(x, kD1, k1), kD3, k3), kD4, k4), kD5, k5), kD6, k6);
  }

  // UT mean: out[b, col] += w[p] * x_end[p, col]
#pragma unroll
  for (int v = 0; v < 8; ++v) {
    int p = tileBase + v + 8 * g;
    if (p < P) {
      float wp = Wp[p];
      atomicAdd(out + (p / S) * NXC + col, wp * x[v]);
    }
  }
}

// ---------------------------------------------------------------------------
extern "C" void kernel_launch(void* const* d_in, const int* in_sizes, int n_in,
                              void* d_out, int out_size, void* d_ws, size_t ws_size,
                              hipStream_t stream) {
  const float* sp = (const float*)d_in[0];   // [B,S,20]
  const float* Wp = (const float*)d_in[1];   // [B,S]
  const float* W1 = (const float*)d_in[2];
  const float* b1 = (const float*)d_in[3];
  const float* W2 = (const float*)d_in[4];
  const float* b2 = (const float*)d_in[5];
  const float* W3 = (const float*)d_in[6];
  const float* b3 = (const float*)d_in[7];

  const int P  = in_sizes[1];                // B*S particles
  const int Bb = out_size / NXC;             // batches
  const int S  = P / Bb;                     // sigma points per batch

  _Float16* wf = (_Float16*)d_ws;
  float*    wb = (float*)((char*)d_ws + (size_t)WF_HALFS * sizeof(_Float16));

  sin_prep_kernel<<<1, 256, 0, stream>>>(W1, W2, W3, b1, b2, b3, wf, wb);
  sin_zero_kernel<<<(out_size + 255) / 256, 256, 0, stream>>>((float*)d_out, out_size);

  const int blocks = (P + 63) / 64;          // 64 particles (4 wave-tiles) per block
  sin_integrate_kernel<<<blocks, 128, 0, stream>>>(sp, Wp, wf, wb, (float*)d_out, P, S);
}
